// GraphFlow_NN_22471268892730
// MI455X (gfx1250) — matrix operand order repaired and checked by reading, not compiled, not run on hardware
//
#include <hip/hip_runtime.h>
#include <hip/hip_bf16.h>

typedef __attribute__((ext_vector_type(16))) _Float16 v16h;
typedef __attribute__((ext_vector_type(8)))  _Float16 v8h;
typedef __attribute__((ext_vector_type(8)))  float    v8f;

union V16 { v16h v; v8h h[2]; };

#define WAVES_PER_BLOCK 8
#define LDS_FENCE() asm volatile("s_wait_dscnt 0x0" ::: "memory")

__device__ __forceinline__ float fast_tanh(float x) {
#if __has_builtin(__builtin_amdgcn_tanhf)
    return __builtin_amdgcn_tanhf(x);   // v_tanh_f32 (gfx1250 trans op)
#else
    return tanhf(x);
#endif
}

__device__ __forceinline__ void cswap(int& a, int& b) {
    int lo = min(a, b), hi = max(a, b);
    a = lo; b = hi;
}

__global__ __launch_bounds__(256)
void graphflow_cnf_kernel(const float* __restrict__ t,
                          const float* __restrict__ data,
                          const int*   __restrict__ edges,   // [2, nodes*4]
                          const float* __restrict__ W1,      // [161,15]
                          const float* __restrict__ b1,      // [15]
                          const float* __restrict__ W2,      // [15,32]
                          const float* __restrict__ b2,      // [32]
                          float*       __restrict__ out,     // [nodes,32]
                          int nodes, int numTiles) {
    // LDS staging: weights/biases (block-shared) + per-wave tiles
    __shared__ alignas(16) _Float16 w1t[16][192];                    // W1^T padded (n,k)
    __shared__ alignas(16) _Float16 w2t[32][32];                     // W2^T padded (n,k)
    __shared__ alignas(16) float    b1s[16];                         // b1 padded
    __shared__ alignas(16) _Float16 x_lds[WAVES_PER_BLOCK][16][192]; // x tile (m,k) f16
    __shared__ alignas(16) _Float16 h_lds[WAVES_PER_BLOCK][16][32];  // h tile (m,k) f16
    __shared__ int nbr_lds[WAVES_PER_BLOCK][16][4];

    const int tid = threadIdx.x;

    // --- Stage W1^T [16][192]: w1t[n][k] = (n<15 && k<161) ? W1[k*15+n] : 0
    for (int i = tid; i < 16 * 192; i += 256) {
        int nn = i / 192, kk = i % 192;
        float v = (nn < 15 && kk < 161) ? W1[kk * 15 + nn] : 0.0f;
        w1t[nn][kk] = (_Float16)v;
    }
    // --- Stage W2^T [32][32]: w2t[n][k] = (k<15) ? W2[k*32+n] : 0
    for (int i = tid; i < 32 * 32; i += 256) {
        int nn = i / 32, kk = i % 32;
        float v = (kk < 15) ? W2[kk * 32 + nn] : 0.0f;
        w2t[nn][kk] = (_Float16)v;
    }
    // --- Stage b1 padded to 16 (col 15 pad must be 0 so tanh(0)=0 keeps K-pad clean)
    if (tid < 16) b1s[tid] = (tid < 15) ? b1[tid] : 0.0f;
    __syncthreads();

    const int wv   = tid >> 5;
    const int lane = tid & 31;
    const int tile = blockIdx.x * WAVES_PER_BLOCK + wv;
    if (tile >= numTiles) return;   // wave-uniform; surviving waves keep EXEC all-ones

    const int base = tile * 16;
    const _Float16 th = (_Float16)t[0];

    // --- t column + zero padding (k = 161..191)
    #pragma unroll
    for (int m = 0; m < 16; ++m) {
        if (lane == 0) x_lds[wv][m][0] = th;
        else           x_lds[wv][m][160 + lane] = (_Float16)0.0f; // lanes 1..31 -> 161..191
    }

    // --- Neighbor ids: dst = edges[1], sorted ascending per node (jnp.sort)
    const int* edges_dst = edges + (size_t)nodes * 4;
    if (lane < 16) {
        int node = base + lane;
        const int4 nv = *(const int4*)(edges_dst + (size_t)node * 4);
        int a0 = nv.x, a1 = nv.y, a2 = nv.z, a3 = nv.w;
        cswap(a0, a1); cswap(a2, a3); cswap(a0, a2); cswap(a1, a3); cswap(a1, a2);
        nbr_lds[wv][lane][0] = a0; nbr_lds[wv][lane][1] = a1;
        nbr_lds[wv][lane][2] = a2; nbr_lds[wv][lane][3] = a3;
    }
    LDS_FENCE();

    // --- Stage features: x = [t | self(32) | nbr0..nbr3(32 each)], coalesced 128B rows
    #pragma unroll
    for (int s = 0; s < 5; ++s) {
        #pragma unroll
        for (int m = 0; m < 16; ++m) {
            int row = (s == 0) ? (base + m) : nbr_lds[wv][m][s - 1];
            float v = data[(size_t)row * 32 + lane];
            x_lds[wv][m][1 + s * 32 + lane] = (_Float16)v;
        }
    }
    LDS_FENCE();

    // --- GEMM1: [16x192] x [192x16] via 6 x v_wmma_f32_16x16x32_f16
    const int half8 = (lane >> 4) << 3;   // unused: A and B both load k = 16h + s via koff
    const int koff  = (lane >> 4) << 4;   // A/B: k offset per lane half
    const int ncd   = lane & 15;          // B/C/D column, A row
    const _Float16* xrow = &x_lds[wv][ncd][0];
    const _Float16* wrow = &w1t[ncd][0];

    v8f acc = {0.f, 0.f, 0.f, 0.f, 0.f, 0.f, 0.f, 0.f};
    #pragma unroll
    for (int c = 0; c < 6; ++c) {
        const int kc = c * 32;
        V16 a, b;
        a.h[0] = *(const v8h*)(xrow + kc + koff);
        a.h[1] = *(const v8h*)(xrow + kc + koff + 8);
        b.h[0] = *(const v8h*)(wrow + kc + koff);
        b.h[1] = *(const v8h*)(wrow + kc + koff + 8);
        acc = __builtin_amdgcn_wmma_f32_16x16x32_f16(false, a.v, false, b.v,
                                                     (short)0, acc, false, false);
    }

    // --- bias + hardware tanh, re-layout D(16x16) -> A(16x32, zero-padded K) via LDS
    const int mbase = (lane >> 4) << 3;
    const float bias1 = b1s[ncd];   // uniform LDS broadcast, zero-padded
    #pragma unroll
    for (int r = 0; r < 8; ++r) {
        float hv = fast_tanh(acc[r] + bias1);      // col 15: tanh(0)=0 -> pad stays 0
        h_lds[wv][mbase + r][ncd]      = (_Float16)hv;
        h_lds[wv][mbase + r][ncd + 16] = (_Float16)0.0f;
    }
    LDS_FENCE();

    // --- GEMM2: [16x32] x [32x32] as two 16x16x32 WMMAs, bias preloaded into C
    const _Float16* hrow = &h_lds[wv][ncd][0];
    V16 ah, b0, b1f;
    ah.h[0]  = *(const v8h*)(hrow + koff);
    ah.h[1]  = *(const v8h*)(hrow + koff + 8);
    const _Float16* w2r0 = &w2t[ncd][0];
    const _Float16* w2r1 = &w2t[16 + ncd][0];
    b0.h[0]  = *(const v8h*)(w2r0 + koff);
    b0.h[1]  = *(const v8h*)(w2r0 + koff + 8);
    b1f.h[0] = *(const v8h*)(w2r1 + koff);
    b1f.h[1] = *(const v8h*)(w2r1 + koff + 8);

    const float bv0 = b2[ncd], bv1 = b2[ncd + 16];
    v8f c0, c1;
    #pragma unroll
    for (int r = 0; r < 8; ++r) { c0[r] = bv0; c1[r] = bv1; }

    v8f d0 = __builtin_amdgcn_wmma_f32_16x16x32_f16(false, ah.v, false, b0.v,
                                                    (short)0, c0, false, false);
    v8f d1 = __builtin_amdgcn_wmma_f32_16x16x32_f16(false, ah.v, false, b1f.v,
                                                    (short)0, c1, false, false);

    // --- Store D: lane covers (m = mbase+r, n = ncd) and (n = ncd+16)
    #pragma unroll
    for (int r = 0; r < 8; ++r) {
        size_t orow = (size_t)(base + mbase + r) * 32;
        out[orow + ncd]      = d0[r];
        out[orow + 16 + ncd] = d1[r];
    }
}

extern "C" void kernel_launch(void* const* d_in, const int* in_sizes, int n_in,
                              void* d_out, int out_size, void* d_ws, size_t ws_size,
                              hipStream_t stream) {
    const float* t     = (const float*)d_in[0];
    const float* data  = (const float*)d_in[1];
    const int*   edges = (const int*)  d_in[2];
    const float* W1    = (const float*)d_in[3];
    const float* b1    = (const float*)d_in[4];
    const float* W2    = (const float*)d_in[5];
    const float* b2    = (const float*)d_in[6];
    float*       out   = (float*)d_out;

    const int nodes    = in_sizes[1] / 32;       // C_IN = 32
    const int numTiles = nodes / 16;             // N divisible by 16 (1,000,000)
    const int blocks   = (numTiles + WAVES_PER_BLOCK - 1) / WAVES_PER_BLOCK;

    graphflow_cnf_kernel<<<blocks, 256, 0, stream>>>(t, data, edges, W1, b1, W2, b2,
                                                     out, nodes, numTiles);
}
